// NO_3762391351825
// MI455X (gfx1250) — matrix operand order repaired and checked by reading, not compiled, not run on hardware
//
#include <hip/hip_runtime.h>
#include <hip/hip_bf16.h>

// ---------------------------------------------------------------------------
// Problem constants (match reference: B=4, N=4096, HEADS=8, HEADC=32, K=8, L=10)
// ---------------------------------------------------------------------------
#define BATCH 4
#define NPTS  4096
#define KNN   8
#define PEL   10

typedef __attribute__((ext_vector_type(16))) _Float16 v16h;
typedef __attribute__((ext_vector_type(8)))  float    v8f;

__device__ __forceinline__ float gelu_exact(float x) {
  return 0.5f * x * (1.0f + erff(x * 0.70710678118654752f));
}

// ---------------------------------------------------------------------------
// Strided GEMM on the WMMA pipe (f16 inputs, f32 accumulate), templated on the
// element strides so every inner-loop load is a global_load with an immediate
// offset from one per-lane base pointer (no per-element address math, no
// spills). Each wave computes a 16 x (16*NT) tile; the A fragment is reused
// across the NT N-subtiles -> NT back-to-back v_wmma per 32-wide K step.
//
//   C[z*cZ + m*ldc + n] = epi( alpha * sum_k A(z,m,k) * B(z,k,n) )
//   A(z,m,k) = A[z*aZ + m*aM + k*AK],  B(z,k,n) = B[z*bZ + k*BK + n*BN]
//   epi: y = alpha*acc; +bias[m]; BN: y = bnG[m]/sqrt(1+eps)*y + bnB[m];
//        +res[z*resZ + m*resLd + n]; act: 0=none, 1=exact GELU, 2=LeakyReLU(0.2)
//
// Per-lane element indices follow the CDNA5 ISA f16 VGPR layouts; the
// half-dependent parts (8*half for both A and B) are folded into the base
// pointers, so remaining per-element offsets are compile-time constants:
//   A 16x32: K = (v>=4?16:0) + 8h + 2(v&3) + j
//   B 32x16: K = (v>=4?16:0) + 8h + 2(v&3) + j ; column = lane&15
//   C/D:     m = tm + r + 8h, n = tn + (lane&15)
//
// A rows with m >= M are CLAMPED to row M-1 (in-bounds garbage); their D rows
// are never stored. N must be a multiple of 16*NT (guaranteed per call site).
// Main loop covers Kd & ~31 unconditionally; one guarded tail handles K % 32.
// ---------------------------------------------------------------------------
template<int AK, int BK, int BN, int NT>
__global__ void gemm_wmma_kernel(
    const float* __restrict__ A, long aM, long aZ,
    const float* __restrict__ Bm, long bZ,
    float* __restrict__ C, long ldc, long cZ,
    int M, int Kd,
    const float* __restrict__ bias,
    const float* __restrict__ bnG, const float* __restrict__ bnB,
    const float* __restrict__ res, long resLd, long resZ,
    float alpha, int act)
{
  const int lane = threadIdx.x;
  const int half = lane >> 4;
  const int l16  = lane & 15;
  const int tn   = blockIdx.x * (16 * NT);
  const int tm   = blockIdx.y * 16;
  const long z   = blockIdx.z;

  A  += z * aZ;
  Bm += z * bZ;
  C  += z * cZ;
  if (res) res += z * resZ;

  const int am  = tm + l16;
  const int amc = (am < M) ? am : (M - 1);          // clamped, always in-bounds
  const float* aBase = A + (long)amc * aM + (long)(8 * half) * AK;
  const float* bBase = Bm + (long)(8 * half) * BK + (long)(tn + l16) * BN;

  v8f acc[NT];
#pragma unroll
  for (int t = 0; t < NT; ++t) acc[t] = v8f{};

  const int kMain = Kd & ~31;

  for (int k0 = 0; k0 < kMain; k0 += 32) {
    if (k0 + 32 < kMain) {  // prefetch next K slab (global_prefetch_b8)
      __builtin_prefetch((const void*)(aBase + (long)32 * AK), 0, 1);
      __builtin_prefetch((const void*)(bBase + (long)32 * BK), 0, 1);
    }
    v16h av;
#pragma unroll
    for (int e = 0; e < 16; ++e) {
      const int v = e >> 1, j = e & 1;
      av[e] = (_Float16)aBase[((v & 4) ? 16 : 0) * AK + (2 * (v & 3) + j) * AK];
    }
#pragma unroll
    for (int t = 0; t < NT; ++t) {
      v16h bv;
#pragma unroll
      for (int e = 0; e < 16; ++e) {
        const int v = e >> 1, j = e & 1;
        bv[e] = (_Float16)bBase[(long)(((v & 4) ? 16 : 0) + 2 * (v & 3) + j) * BK + (long)(16 * t) * BN];
      }
      acc[t] = __builtin_amdgcn_wmma_f32_16x16x32_f16(false, av, false, bv,
                                                      (short)0, acc[t], false, false);
    }
    aBase += (long)32 * AK;
    bBase += (long)32 * BK;
  }

  if (kMain < Kd) {  // guarded K tail (one step; K = 51, 319, 3, ...)
    v16h av = {};
#pragma unroll
    for (int e = 0; e < 16; ++e) {
      const int v = e >> 1, j = e & 1;
      const int ka = kMain + ((v & 4) ? 16 : 0) + 8 * half + 2 * (v & 3) + j;
      if (ka < Kd)
        av[e] = (_Float16)aBase[((v & 4) ? 16 : 0) * AK + (2 * (v & 3) + j) * AK];
    }
#pragma unroll
    for (int t = 0; t < NT; ++t) {
      v16h bv = {};
#pragma unroll
      for (int e = 0; e < 16; ++e) {
        const int v = e >> 1, j = e & 1;
        const int kb = kMain + ((v & 4) ? 16 : 0) + 8 * half + 2 * (v & 3) + j;
        if (kb < Kd)
          bv[e] = (_Float16)bBase[(long)(((v & 4) ? 16 : 0) + 2 * (v & 3) + j) * BK + (long)(16 * t) * BN];
      }
      acc[t] = __builtin_amdgcn_wmma_f32_16x16x32_f16(false, av, false, bv,
                                                      (short)0, acc[t], false, false);
    }
  }

  // Epilogue (N is a multiple of 16*NT -> no column guard)
  const float invs = rsqrtf(1.0f + 1e-5f);   // eval-mode BN: g/sqrt(1+eps)
#pragma unroll
  for (int t = 0; t < NT; ++t) {
    const int bn = tn + 16 * t + l16;
#pragma unroll
    for (int r = 0; r < 8; ++r) {
      const int m = tm + r + 8 * half;
      if (m >= M) continue;
      float y = alpha * acc[t][r];
      if (bias) y += bias[m];
      if (bnG)  y = bnG[m] * invs * y + bnB[m];
      if (res)  y += res[(long)m * resLd + bn];
      if (act == 1)      y = gelu_exact(y);
      else if (act == 2) y = (y >= 0.0f) ? y : 0.2f * y;
      C[(long)m * ldc + bn] = y;
    }
  }
}

template<int AK, int BK, int BN, int NT>
static inline void launch_gemm(hipStream_t stream,
    const float* A, long aM, long aZ,
    const float* Bm, long bZ,
    float* C, long ldc, long cZ,
    int M, int Kd, int Nn, int Z,
    const float* bias, const float* bnG, const float* bnB,
    const float* res, long resLd, long resZ,
    float alpha, int act)
{
  dim3 g(Nn / (16 * NT), (M + 15) / 16, Z);   // Nn % (16*NT) == 0 by construction
  gemm_wmma_kernel<AK, BK, BN, NT><<<g, dim3(32), 0, stream>>>(
      A, aM, aZ, Bm, bZ, C, ldc, cZ, M, Kd,
      bias, bnG, bnB, res, resLd, resZ, alpha, act);
}

// ---------------------------------------------------------------------------
// Lifting features: feat[b, 0..50, n] = [sin(proj 24), cos(proj 24), x 3]
// proj channel c*8+o = x_c * 2^o * pi
// ---------------------------------------------------------------------------
__global__ void lift_feat_kernel(const float* __restrict__ x, float* __restrict__ feat)
{
  int t = blockIdx.x * blockDim.x + threadIdx.x;
  if (t >= BATCH * NPTS) return;
  const int n = t % NPTS, b = t / NPTS;
  const float pi = 3.14159265358979323846f;
  float pv[3];
#pragma unroll
  for (int c = 0; c < 3; ++c) pv[c] = x[((long)b * 3 + c) * NPTS + n];
  float* f = feat + (long)b * 51 * NPTS + n;
#pragma unroll
  for (int c = 0; c < 3; ++c) {
    float sc = pi;
#pragma unroll
    for (int o = 0; o < 8; ++o) {
      const float pr = pv[c] * sc;
      f[(long)(c * 8 + o) * NPTS]      = sinf(pr);
      f[(long)(24 + c * 8 + o) * NPTS] = cosf(pr);
      sc *= 2.0f;
    }
    f[(long)(48 + c) * NPTS] = pv[c];
  }
}

// ---------------------------------------------------------------------------
// Per-(b,h,n) LayerNorm with unbiased std over the 32 head-channels of k and v.
// qkv layout [B,768,N]; k at channel 96h+32+c, v at 96h+64+c. In-place.
// ---------------------------------------------------------------------------
__global__ void tln_kernel(float* __restrict__ qkv,
                           const float* __restrict__ kg, const float* __restrict__ kb,
                           const float* __restrict__ vg, const float* __restrict__ vb)
{
  int t = blockIdx.x * blockDim.x + threadIdx.x;
  if (t >= BATCH * 8 * NPTS) return;
  const int n = t % NPTS;
  const int bh = t / NPTS;
  const int h = bh & 7, b = bh >> 3;
  const long base = ((long)b * 768 + 96L * h) * NPTS + n;
  for (int which = 0; which < 2; ++which) {
    float* p = qkv + base + (which ? 64L : 32L) * NPTS;
    const float* g  = (which ? vg : kg) + h * 32;
    const float* bb = (which ? vb : kb) + h * 32;
    float vals[32], m = 0.0f;
#pragma unroll
    for (int c = 0; c < 32; ++c) { vals[c] = p[(long)c * NPTS]; m += vals[c]; }
    m *= (1.0f / 32.0f);
    float var = 0.0f;
#pragma unroll
    for (int c = 0; c < 32; ++c) { const float d = vals[c] - m; var += d * d; }
    const float s = sqrtf(var * (1.0f / 31.0f));       // ddof=1
    const float inv = 1.0f / (s + 1e-5f);
#pragma unroll
    for (int c = 0; c < 32; ++c)
      p[(long)c * NPTS] = g[c] * ((vals[c] - m) * inv) + bb[c];
  }
}

// ---------------------------------------------------------------------------
// Exact KNN (K=8, includes self) with LDS point tiles; emits
// local[b,c,n,k] = p_n[c] - p_knn[c]. d2 = |p|^2 + |q|^2 - 2 p.q (as reference).
// Stable top-8: strict-less insertion keeps earlier index first on ties,
// matching jax.lax.top_k(-d2).
// ---------------------------------------------------------------------------
__global__ void knn_local_kernel(const float* __restrict__ pc, float* __restrict__ local)
{
  __shared__ float sx[256], sy[256], sz[256], sq[256];
  const int tid = threadIdx.x;
  const int n = blockIdx.x * 256 + tid;
  const int b = blockIdx.y;
  const float* px = pc + (long)b * 3 * NPTS;
  const float myx = px[n], myy = px[NPTS + n], myz = px[2L * NPTS + n];
  const float mys = myx * myx + myy * myy + myz * myz;
  float bd[KNN]; int bi[KNN];
#pragma unroll
  for (int k = 0; k < KNN; ++k) { bd[k] = 3.4e38f; bi[k] = 0; }
  for (int t0 = 0; t0 < NPTS; t0 += 256) {
    __syncthreads();
    sx[tid] = px[t0 + tid];
    sy[tid] = px[NPTS + t0 + tid];
    sz[tid] = px[2L * NPTS + t0 + tid];
    sq[tid] = sx[tid] * sx[tid] + sy[tid] * sy[tid] + sz[tid] * sz[tid];
    __syncthreads();
    for (int m = 0; m < 256; ++m) {
      const float dot = myx * sx[m] + myy * sy[m] + myz * sz[m];
      const float d2 = mys + sq[m] - 2.0f * dot;
      if (d2 < bd[KNN - 1]) {
        int j = KNN - 1;
        while (j > 0 && bd[j - 1] > d2) { bd[j] = bd[j - 1]; bi[j] = bi[j - 1]; --j; }
        bd[j] = d2; bi[j] = t0 + m;
      }
    }
  }
  float* lp = local + (long)b * 3 * NPTS * KNN + (long)n * KNN;
#pragma unroll
  for (int k = 0; k < KNN; ++k) {
    const int idx = bi[k];
    lp[k]                         = myx - px[idx];
    lp[(long)NPTS * KNN + k]      = myy - px[NPTS + idx];
    lp[2L * (long)NPTS * KNN + k] = myz - px[2L * NPTS + idx];
  }
}

// ---------------------------------------------------------------------------
// Pool over K: fmax -> h319 channels 63..190, fmean -> channels 191..318
// ---------------------------------------------------------------------------
__global__ void pool_kernel(const float* __restrict__ f, float* __restrict__ h319)
{
  int t = blockIdx.x * blockDim.x + threadIdx.x;
  if (t >= BATCH * 128 * NPTS) return;
  const int n = t % NPTS;
  const int rest = t / NPTS;
  const int ch = rest % 128, b = rest / 128;
  const float* p = f + ((long)b * 128 + ch) * (long)(NPTS * KNN) + (long)n * KNN;
  float mx = p[0], sm = p[0];
#pragma unroll
  for (int k = 1; k < KNN; ++k) { const float v = p[k]; mx = fmaxf(mx, v); sm += v; }
  h319[((long)b * 319 + 63 + ch) * NPTS + n]  = mx;
  h319[((long)b * 319 + 191 + ch) * NPTS + n] = sm * (1.0f / KNN);
}

// ---------------------------------------------------------------------------
// Positional encoding into h319 channels 0..62: [pc (3), sin(pc*2^l) (30), cos (30)]
// pe channel = c*10 + l
// ---------------------------------------------------------------------------
__global__ void enc_kernel(const float* __restrict__ pc, float* __restrict__ h319)
{
  int t = blockIdx.x * blockDim.x + threadIdx.x;
  if (t >= BATCH * NPTS) return;
  const int n = t % NPTS, b = t / NPTS;
  float* hp = h319 + (long)b * 319 * NPTS + n;
#pragma unroll
  for (int c = 0; c < 3; ++c) {
    const float v = pc[((long)b * 3 + c) * NPTS + n];
    hp[(long)c * NPTS] = v;
    float sc = 1.0f;
#pragma unroll
    for (int l = 0; l < PEL; ++l) {
      const float pe = v * sc;
      hp[(long)(3 + c * PEL + l) * NPTS]  = sinf(pe);
      hp[(long)(33 + c * PEL + l) * NPTS] = cosf(pe);
      sc *= 2.0f;
    }
  }
}

// ---------------------------------------------------------------------------
// out[b,n,c] = pc_out[b,c,n] + dh[b,c,n] + x[b,c,n]
// ---------------------------------------------------------------------------
__global__ void final_kernel(const float* __restrict__ pcb, const float* __restrict__ dh,
                             const float* __restrict__ x, float* __restrict__ out)
{
  int t = blockIdx.x * blockDim.x + threadIdx.x;
  if (t >= BATCH * NPTS) return;
  const int n = t % NPTS, b = t / NPTS;
#pragma unroll
  for (int c = 0; c < 3; ++c) {
    const long off = ((long)b * 3 + c) * NPTS + n;
    out[((long)b * NPTS + n) * 3 + c] = pcb[off] + dh[off] + x[off];
  }
}

// ---------------------------------------------------------------------------
// Host-side orchestration.
// Input pointer order = setup_inputs() dict order:
//   0:x, 1:lift_w, 2:lift_b,
//   per block i in {1,2} (base = 3 + 10*(i-1)):
//     +0 qkv_w, +1 qkv_b, +2 o1_w, +3 o1_b, +4 o2_w, +5 o2_b,
//     +6 kln_g, +7 kln_b, +8 vln_g, +9 vln_b
//   23:out_w1 24:out_b1 25:out_w2 26:out_b2
//   27..34: pw0,pb0,pw1,pb1,pw2,pb2,pw3,pb3
//   35..40: pbn_g0,pbn_b0,pbn_g1,pbn_b1,pbn_g2,pbn_b2
//   41..48: dw0,db0,dw1,db1,dw2,db2,dw3,db3
//   49..54: dbn_g0,dbn_b0,dbn_g1,dbn_b1,dbn_g2,dbn_b2
// ---------------------------------------------------------------------------
extern "C" void kernel_launch(void* const* d_in, const int* in_sizes, int n_in,
                              void* d_out, int out_size, void* d_ws, size_t ws_size,
                              hipStream_t stream)
{
  (void)in_sizes; (void)n_in; (void)out_size; (void)ws_size;
  const float* x = (const float*)d_in[0];
  auto P = [&](int i) { return (const float*)d_in[i]; };

  const long NN = (long)NPTS;
  const long NK = NN * KNN;

  float* W = (float*)d_ws;
  size_t off = 0;
  auto alloc = [&](long nf) { float* p = W + off; off += (size_t)nf; return p; };
  float* H0    = alloc(4 * 256 * NN);
  float* H1    = alloc(4 * 256 * NN);
  float* H2    = alloc(4 * 256 * NN);
  float* QKV   = alloc(4 * 768 * NN);
  float* KVB   = alloc(4 * 8 * 32 * 32);
  float* FEAT  = alloc(4 * 51 * NN);
  float* OUT1  = alloc(4 * 128 * NN);
  float* PCB   = alloc(4 * 3 * NN);
  float* LOCAL = alloc(4 * 3 * NK);
  float* C64   = alloc(4 * 64 * NK);
  float* C128A = alloc(4 * 128 * NK);
  float* C128B = alloc(4 * 128 * NK);
  float* H319  = alloc(4 * 319 * NN);
  float* D256  = alloc(4 * 256 * NN);
  float* D128  = alloc(4 * 128 * NN);
  float* D64   = alloc(4 * 64 * NN);
  float* DH    = alloc(4 * 3 * NN);
  // requires ws_size >= off*sizeof(float) (~333 MB); fits easily in 432 GiB HBM.

  const int tBN = BATCH * NPTS;

  // ---- Lifting: feat(51) then H0 = lift_w^T @ feat + lift_b ----------------
  lift_feat_kernel<<<(tBN + 255) / 256, 256, 0, stream>>>(x, FEAT);
  launch_gemm<256, 4096, 1, 4>(stream,
      P(1), 1, 0,                       // A(m=o,k=d) = lift_w[d*256+o]
      FEAT, 51 * NN,
      H0, NN, 256 * NN,
      256, 51, NPTS, BATCH,
      P(2), nullptr, nullptr, nullptr, 0, 0, 1.0f, 0);

  // ---- Two Galerkin attention blocks --------------------------------------
  float* hin = H0;
  for (int blk = 0; blk < 2; ++blk) {
    const int base = 3 + blk * 10;
    const float *qkv_w = P(base + 0), *qkv_b = P(base + 1);
    const float *o1w = P(base + 2), *o1b = P(base + 3);
    const float *o2w = P(base + 4), *o2b = P(base + 5);
    const float *kg = P(base + 6), *kb = P(base + 7);
    const float *vg = P(base + 8), *vb = P(base + 9);
    float* ret = (blk == 0) ? H1 : H0;

    // qkv = W_qkv @ h + b  -> [B,768,N]
    launch_gemm<1, 4096, 1, 4>(stream,
        qkv_w, 256, 0, hin, 256 * NN,
        QKV, NN, 768 * NN, 768, 256, NPTS, BATCH,
        qkv_b, nullptr, nullptr, nullptr, 0, 0, 1.0f, 0);

    // LayerNorm on k and v (in place)
    tln_kernel<<<(BATCH * 8 * NPTS + 255) / 256, 256, 0, stream>>>(QKV, kg, kb, vg, vb);

    // kv[b,h,c,d] = sum_n k*v / N   (z = b*8+h, stride 96*NN works since 768=8*96)
    launch_gemm<1, 1, 4096, 2>(stream,
        QKV + 32 * NN, NN, 96 * NN,     // A(c,n) = k
        QKV + 64 * NN, 96 * NN,         // B(n,d) = v[d*NN + n]
        KVB, 32, 1024,
        32, NPTS, 32, BATCH * 8,
        nullptr, nullptr, nullptr, nullptr, 0, 0, 1.0f / NPTS, 0);

    // ret = q @ kv + hin   (A(m=d,k=c) = kv[c*32+d]; out channel 32h+d)
    launch_gemm<32, 4096, 1, 4>(stream,
        KVB, 1, 1024,
        QKV, 96 * NN,
        ret, NN, 32 * NN,
        32, 32, NPTS, BATCH * 8,
        nullptr, nullptr, nullptr, hin, NN, 32 * NN, 1.0f, 0);

    // h1 = gelu(o1 @ ret + b)
    launch_gemm<1, 4096, 1, 4>(stream,
        o1w, 256, 0, ret, 256 * NN,
        H2, NN, 256 * NN, 256, 256, NPTS, BATCH,
        o1b, nullptr, nullptr, nullptr, 0, 0, 1.0f, 1);

    // h_out = o2 @ h1 + b + hin
    launch_gemm<1, 4096, 1, 4>(stream,
        o2w, 256, 0, H2, 256 * NN,
        ret, NN, 256 * NN, 256, 256, NPTS, BATCH,
        o2b, nullptr, nullptr, hin, NN, 256 * NN, 1.0f, 0);

    hin = ret;
  }

  // ---- to_outputs: 256 -> 128 (GELU) -> 3 ---------------------------------
  launch_gemm<1, 4096, 1, 4>(stream,
      P(23), 256, 0, hin, 256 * NN,
      OUT1, NN, 128 * NN, 128, 256, NPTS, BATCH,
      P(24), nullptr, nullptr, nullptr, 0, 0, 1.0f, 1);
  launch_gemm<1, 4096, 1, 4>(stream,
      P(25), 128, 0, OUT1, 128 * NN,
      PCB, NN, 3 * NN, 3, 128, NPTS, BATCH,
      P(26), nullptr, nullptr, nullptr, 0, 0, 1.0f, 0);

  // ---- DisplaceNet --------------------------------------------------------
  knn_local_kernel<<<dim3(NPTS / 256, BATCH), 256, 0, stream>>>(PCB, LOCAL);

  // patch feature net over N*K columns, BN folded into epilogue, LeakyReLU(0.2)
  launch_gemm<1, 32768, 1, 4>(stream,
      P(27), 3, 0, LOCAL, 3 * NK,
      C64, NK, 64 * NK, 64, 3, (int)NK, BATCH,
      P(28), P(35), P(36), nullptr, 0, 0, 1.0f, 2);
  launch_gemm<1, 32768, 1, 4>(stream,
      P(29), 64, 0, C64, 64 * NK,
      C128A, NK, 128 * NK, 128, 64, (int)NK, BATCH,
      P(30), P(37), P(38), nullptr, 0, 0, 1.0f, 2);
  launch_gemm<1, 32768, 1, 4>(stream,
      P(31), 128, 0, C128A, 128 * NK,
      C128B, NK, 128 * NK, 128, 128, (int)NK, BATCH,
      P(32), P(39), P(40), nullptr, 0, 0, 1.0f, 2);
  launch_gemm<1, 32768, 1, 4>(stream,
      P(33), 128, 0, C128B, 128 * NK,
      C128A, NK, 128 * NK, 128, 128, (int)NK, BATCH,
      P(34), nullptr, nullptr, nullptr, 0, 0, 1.0f, 0);

  pool_kernel<<<(BATCH * 128 * NPTS + 255) / 256, 256, 0, stream>>>(C128A, H319);
  enc_kernel<<<(tBN + 255) / 256, 256, 0, stream>>>(PCB, H319);

  // displace_net: 319 -> 256 -> 128 -> 64 -> 3 (BN + lrelu on first 3)
  launch_gemm<1, 4096, 1, 4>(stream,
      P(41), 319, 0, H319, 319 * NN,
      D256, NN, 256 * NN, 256, 319, NPTS, BATCH,
      P(42), P(49), P(50), nullptr, 0, 0, 1.0f, 2);
  launch_gemm<1, 4096, 1, 4>(stream,
      P(43), 256, 0, D256, 256 * NN,
      D128, NN, 128 * NN, 128, 256, NPTS, BATCH,
      P(44), P(51), P(52), nullptr, 0, 0, 1.0f, 2);
  launch_gemm<1, 4096, 1, 4>(stream,
      P(45), 128, 0, D128, 128 * NN,
      D64, NN, 64 * NN, 64, 128, NPTS, BATCH,
      P(46), P(53), P(54), nullptr, 0, 0, 1.0f, 2);
  launch_gemm<1, 4096, 1, 4>(stream,
      P(47), 64, 0, D64, 64 * NN,
      DH, NN, 3 * NN, 3, 64, NPTS, BATCH,
      P(48), nullptr, nullptr, nullptr, 0, 0, 1.0f, 0);

  // ---- final: out[b,n,c] = pc_out + dh + x_orig ---------------------------
  final_kernel<<<(tBN + 255) / 256, 256, 0, stream>>>(PCB, DH, x, (float*)d_out);
}
